// KERN_17575006175920
// MI455X (gfx1250) — matrix operand order repaired and checked by reading, not compiled, not run on hardware
//
#include <hip/hip_runtime.h>
#include <hip/hip_bf16.h>

// Problem constants (match reference)
#define Bc 8
#define Nc 64
#define Rc 2048
#define Dc 4096
#define Hc 512
#define NOc 151
#define NRc 51
#define Tc 3
#define Pc (Bc * Nc)   // 512 object rows
#define Qc (Bc * Rc)   // 16384 relation rows

typedef __attribute__((ext_vector_type(16))) __bf16 v16bf;
typedef __attribute__((ext_vector_type(8)))  float  v8f;
typedef __attribute__((ext_vector_type(4)))  unsigned int v4u;

static __device__ __forceinline__ unsigned int f32_to_bf16_rne(float f) {
    unsigned u = __float_as_uint(f);
    unsigned r = u + 0x7FFFu + ((u >> 16) & 1u);
    return r >> 16;
}

// ---------------------------------------------------------------------------
// Generic GEMM: C[M,N] (= or +=) A[M,K] @ B[K,N] (+ bias[N])
// fp32 in/out, bf16 WMMA compute. 128x64 C tile per 256-thread workgroup.
// 8 waves: wave (wid&3) -> two 16-row bands (2*waveM, 2*waveM+1),
//          wave (wid>>2) -> 32-col band. 4 accumulators/wave:
//          2 A frags x 2 B frags = 4 WMMAs per K-step.
//
// LDS tiles are stored in *fragment order*: lane L of each band owns 16
// contiguous bf16 (32 B, 32B-aligned) laid out exactly as the WMMA VGPR
// layout expects, so a fragment load is 2x ds_load_b128.
//   A (16x32, band b): slot (b*32 + lane)*16 + j, element (row = b*16 + lane%16,
//     K = (j>=8?16:0) + (lane>=16?8:0) + (j&7))
//   B (32x16, colband cb): slot (cb*32 + lane)*16 + j, element
//     (K = (j>=8?16:0) + (lane>=16?8:0) + (j&7), col = cb*16 + lane&15)
// ---------------------------------------------------------------------------
template <bool FULL>
__global__ __launch_bounds__(256) void gemm_wmma_bf16(
    const float* __restrict__ A, int lda,
    const float* __restrict__ Bm, int ldb,
    const float* __restrict__ bias,
    float* __restrict__ C, int ldc,
    int M, int N, int K, int accumulate)
{
    __shared__ __align__(16) unsigned short sA[2][128 * 32];
    __shared__ __align__(16) unsigned short sB[2][32 * 64];

    const int tid   = threadIdx.x;
    const int lane  = tid & 31;
    const int wid   = tid >> 5;
    const int waveM = wid & 3;        // two 16-row bands: 2*waveM, 2*waveM+1
    const int waveN = wid >> 2;       // 0..1  (32-col band)
    const int m0 = blockIdx.y * 128;
    const int n0 = blockIdx.x * 64;

    v8f acc00 = {}, acc01 = {}, acc10 = {}, acc11 = {};

    const int nk = (K + 31) >> 5;

    // -------- persistent per-thread staging state (full path) --------
    const float* aPtr[8];
    const float* bPtr[4];
    int dAoff[8];
    int dBoff[4];
    if (FULL) {
        #pragma unroll
        for (int t = 0; t < 8; ++t) {           // A: 2048 bf16-pairs
            int p = tid + t * 256;
            int r = p >> 4;                     // 0..127
            int c = (p & 15) << 1;              // 0..30, even
            aPtr[t] = A + (long)(m0 + r) * lda + c;
            int band = r >> 4, rl = r & 15;
            int lane2 = (((c >> 3) & 1) << 4) + rl;
            int j = (((c >> 4) & 1) << 3) + (c & 7);
            dAoff[t] = ((band * 32 + lane2) << 4) + j;
        }
        #pragma unroll
        for (int t = 0; t < 4; ++t) {           // B: 1024 pairs along K
            int p = tid + t * 256;
            int k = (p >> 6) << 1;              // 0..30, even
            int c = p & 63;
            bPtr[t] = Bm + (long)k * ldb + (n0 + c);
            int cb = c >> 4;
            int lane2 = (((k >> 3) & 1) << 4) + (c & 15);
            dBoff[t] = ((cb * 32 + lane2) << 4) + (((k >> 4) & 1) << 3) + (k & 7);
        }
    }

    // ---- tile staging ----
    auto stage = [&](int bufIdx, int k0) {
        unsigned short* dA = &sA[bufIdx][0];
        unsigned short* dB = &sB[bufIdx][0];
        if (FULL) {
            #pragma unroll
            for (int t = 0; t < 8; ++t) {
                float2 f = *(const float2*)aPtr[t];
                aPtr[t] += 32;
                *(unsigned*)&dA[dAoff[t]] =
                    f32_to_bf16_rne(f.x) | (f32_to_bf16_rne(f.y) << 16);
            }
            #pragma unroll
            for (int t = 0; t < 4; ++t) {
                float f0 = bPtr[t][0];
                float f1 = bPtr[t][ldb];
                bPtr[t] += (long)ldb * 32;
                *(unsigned*)&dB[dBoff[t]] =
                    f32_to_bf16_rne(f0) | (f32_to_bf16_rne(f1) << 16);
            }
        } else {
            // Guarded path: clamped unconditional loads + cndmask zeroing
            #pragma unroll
            for (int t = 0; t < 16; ++t) {
                int e = tid + t * 256;
                int r = e >> 5, c = e & 31;
                int gm = m0 + r, gk = k0 + c;
                int gmc = gm < (M - 1) ? gm : (M - 1);
                int gkc = gk < (K - 1) ? gk : (K - 1);
                float v = A[(long)gmc * lda + gkc];
                if (gm >= M || gk >= K) v = 0.0f;
                int band = r >> 4, rl = r & 15;
                int lane2 = (((c >> 3) & 1) << 4) + rl;
                int j = (((c >> 4) & 1) << 3) + (c & 7);
                dA[((band * 32 + lane2) << 4) + j] = (unsigned short)f32_to_bf16_rne(v);
            }
            #pragma unroll
            for (int t = 0; t < 8; ++t) {
                int e = tid + t * 256;
                int k = e >> 6, c = e & 63;
                int gk = k0 + k, gn = n0 + c;
                int gkc = gk < (K - 1) ? gk : (K - 1);
                int gnc = gn < (N - 1) ? gn : (N - 1);
                float v = Bm[(long)gkc * ldb + gnc];
                if (gk >= K || gn >= N) v = 0.0f;
                int cb = c >> 4;
                int lane2 = (((k >> 3) & 1) << 4) + (c & 15);
                dB[((cb * 32 + lane2) << 4) + (((k >> 4) & 1) << 3) + (k & 7)] = (unsigned short)f32_to_bf16_rne(v);
            }
        }
    };

    stage(0, 0);
    __syncthreads();

    int buf = 0;
    for (int kt = 0; kt < nk; ++kt) {
        if (kt + 1 < nk) stage(buf ^ 1, (kt + 1) << 5);   // overlap with compute

        union Frag { v16bf v; v4u q[2]; };
        Frag fa0, fa1, fb0, fb1;
        const v4u* pA0 = (const v4u*)&sA[buf][((2 * waveM) * 32 + lane) << 4];
        const v4u* pA1 = (const v4u*)&sA[buf][((2 * waveM + 1) * 32 + lane) << 4];
        const v4u* pB0 = (const v4u*)&sB[buf][((waveN * 2) * 32 + lane) << 4];
        const v4u* pB1 = (const v4u*)&sB[buf][((waveN * 2 + 1) * 32 + lane) << 4];
        fa0.q[0] = pA0[0]; fa0.q[1] = pA0[1];
        fa1.q[0] = pA1[0]; fa1.q[1] = pA1[1];
        fb0.q[0] = pB0[0]; fb0.q[1] = pB0[1];
        fb1.q[0] = pB1[0]; fb1.q[1] = pB1[1];

        acc00 = __builtin_amdgcn_wmma_f32_16x16x32_bf16(false, fa0.v, false, fb0.v,
                                                        (short)0, acc00, false, false);
        acc01 = __builtin_amdgcn_wmma_f32_16x16x32_bf16(false, fa0.v, false, fb1.v,
                                                        (short)0, acc01, false, false);
        acc10 = __builtin_amdgcn_wmma_f32_16x16x32_bf16(false, fa1.v, false, fb0.v,
                                                        (short)0, acc10, false, false);
        acc11 = __builtin_amdgcn_wmma_f32_16x16x32_bf16(false, fa1.v, false, fb1.v,
                                                        (short)0, acc11, false, false);
        __syncthreads();
        buf ^= 1;
    }

    // C/D layout: VGPR r -> M = r + (lane>=16?8:0); N = lane&15
    const int laneHi = (lane >> 4) & 1;
    const int c0 = n0 + waveN * 32 + (lane & 15);
    const int c1 = c0 + 16;
    #pragma unroll
    for (int sub = 0; sub < 2; ++sub) {
        const v8f& a0 = sub ? acc10 : acc00;
        const v8f& a1 = sub ? acc11 : acc01;
        const int rowBase = m0 + (2 * waveM + sub) * 16 + laneHi * 8;
        #pragma unroll
        for (int r = 0; r < 8; ++r) {
            int row = rowBase + r;
            float v0 = a0[r], v1 = a1[r];
            if (FULL) {
                if (bias) { v0 += bias[c0]; v1 += bias[c1]; }
                float* p0 = C + (long)row * ldc + c0;
                float* p1 = C + (long)row * ldc + c1;
                if (accumulate) { *p0 += v0; *p1 += v1; }
                else            { *p0 = v0;  *p1 = v1;  }
            } else {
                if (row >= M) continue;
                if (bias) {
                    if (c0 < N) v0 += bias[c0];
                    if (c1 < N) v1 += bias[c1];
                }
                if (accumulate) {
                    if (c0 < N) C[(long)row * ldc + c0] += v0;
                    if (c1 < N) C[(long)row * ldc + c1] += v1;
                } else {
                    if (c0 < N) C[(long)row * ldc + c0] = v0;
                    if (c1 < N) C[(long)row * ldc + c1] = v1;
                }
            }
        }
    }
}

// ---------------------------------------------------------------------------
// Elementwise / reduction kernels
// ---------------------------------------------------------------------------
__global__ void zero_kernel(float* __restrict__ p, long n) {
    long i = (long)blockIdx.x * blockDim.x + threadIdx.x;
    if (i < n) p[i] = 0.0f;
}

__global__ void tanh_inplace_kernel(float* __restrict__ p, long n) {
    long i = (long)blockIdx.x * blockDim.x + threadIdx.x;
    if (i < n) p[i] = tanhf(p[i]);
}

// gz <- sigmoid(gz); gr <- sigmoid(gr) * h
__global__ void gru_phase1_kernel(float* __restrict__ gz, float* __restrict__ gr,
                                  const float* __restrict__ h, long n) {
    long i = (long)blockIdx.x * blockDim.x + threadIdx.x;
    if (i < n) {
        gz[i] = 1.0f / (1.0f + expf(-gz[i]));
        gr[i] = (1.0f / (1.0f + expf(-gr[i]))) * h[i];
    }
}

// h <- (1-z)*h + z*tanh(gh)
__global__ void gru_phase2_kernel(float* __restrict__ h, const float* __restrict__ z,
                                  const float* __restrict__ gh, long n) {
    long i = (long)blockIdx.x * blockDim.x + threadIdx.x;
    if (i < n) {
        float zz = z[i];
        h[i] = (1.0f - zz) * h[i] + zz * tanhf(gh[i]);
    }
}

// mr[i,:] = Ts[s_i,:] + To[o_i,:]   (indices are global object row ids)
__global__ void gather_kernel(const int* __restrict__ rel_inds,
                              const float* __restrict__ Ts, const float* __restrict__ To,
                              float* __restrict__ mr, int Q, int H) {
    long idx = (long)blockIdx.x * blockDim.x + threadIdx.x;
    long n = (long)Q * H;
    if (idx < n) {
        int i = (int)(idx / H);
        int j = (int)(idx - (long)i * H);
        int s = rel_inds[i * 3 + 1];
        int o = rel_inds[i * 3 + 2];
        mr[idx] = Ts[(long)s * H + j] + To[(long)o * H + j];
    }
}

// Ss[s_i,:] += hr[i,:]; So[o_i,:] += hr[i,:]
__global__ void scatter_kernel(const int* __restrict__ rel_inds,
                               const float* __restrict__ hr,
                               float* __restrict__ Ss, float* __restrict__ So,
                               int Q, int H) {
    long idx = (long)blockIdx.x * blockDim.x + threadIdx.x;
    long n = (long)Q * H;
    if (idx < n) {
        int i = (int)(idx / H);
        int j = (int)(idx - (long)i * H);
        int s = rel_inds[i * 3 + 1];
        int o = rel_inds[i * 3 + 2];
        float v = hr[idx];
        atomicAdd(&Ss[(long)s * H + j], v);
        atomicAdd(&So[(long)o * H + j], v);
    }
}

// One wave32 per row softmax
__global__ void softmax_rows_kernel(const float* __restrict__ x, float* __restrict__ y,
                                    int rows, int cols) {
    int row = blockIdx.x;
    int lane = threadIdx.x;
    if (row >= rows) return;
    const float* xr = x + (long)row * cols;
    float m = -3.4e38f;
    for (int c = lane; c < cols; c += 32) m = fmaxf(m, xr[c]);
    #pragma unroll
    for (int off = 16; off > 0; off >>= 1) m = fmaxf(m, __shfl_xor(m, off, 32));
    float s = 0.0f;
    for (int c = lane; c < cols; c += 32) s += expf(xr[c] - m);
    #pragma unroll
    for (int off = 16; off > 0; off >>= 1) s += __shfl_xor(s, off, 32);
    float inv = 1.0f / s;
    float* yr = y + (long)row * cols;
    for (int c = lane; c < cols; c += 32) yr[c] = expf(xr[c] - m) * inv;
}

// obj_preds = argmax over cols [1, cols) (ties -> lowest index), written as float
__global__ void argmax_kernel(const float* __restrict__ x, float* __restrict__ preds,
                              int rows, int cols) {
    int row = blockIdx.x;
    int lane = threadIdx.x;
    if (row >= rows) return;
    const float* xr = x + (long)row * cols;
    float best = -3.4e38f;
    int bidx = cols;
    for (int c = 1 + lane; c < cols; c += 32) {
        float v = xr[c];
        if (v > best || (v == best && c < bidx)) { best = v; bidx = c; }
    }
    #pragma unroll
    for (int off = 16; off > 0; off >>= 1) {
        float ov = __shfl_xor(best, off, 32);
        int   oi = __shfl_xor(bidx, off, 32);
        if (ov > best || (ov == best && oi < bidx)) { best = ov; bidx = oi; }
    }
    if (lane == 0) preds[row] = (float)bidx;
}

// ---------------------------------------------------------------------------
// Host-side orchestration
// ---------------------------------------------------------------------------
static inline void launch_gemm(const float* A, int lda, const float* Bm, int ldb,
                               const float* bias, float* C, int ldc,
                               int M, int N, int K, int accumulate, hipStream_t s) {
    dim3 grid((N + 63) / 64, (M + 127) / 128);
    bool full = (M % 128 == 0) && (N % 64 == 0) && (K % 32 == 0);
    if (full)
        gemm_wmma_bf16<true><<<grid, 256, 0, s>>>(A, lda, Bm, ldb, bias, C, ldc, M, N, K, accumulate);
    else
        gemm_wmma_bf16<false><<<grid, 256, 0, s>>>(A, lda, Bm, ldb, bias, C, ldc, M, N, K, accumulate);
}

extern "C" void kernel_launch(void* const* d_in, const int* in_sizes, int n_in,
                              void* d_out, int out_size, void* d_ws, size_t ws_size,
                              hipStream_t stream) {
    (void)in_sizes; (void)n_in; (void)out_size; (void)ws_size;

    // Inputs (setup_inputs dict order)
    const int*   rel_inds  = (const int*)  d_in[1];
    const float* obj_fmaps = (const float*)d_in[2];
    const float* obj_logits= (const float*)d_in[3];
    const float* vr        = (const float*)d_in[4];
    const float* W_obj = (const float*)d_in[5];   const float* b_obj = (const float*)d_in[6];
    const float* W_rel = (const float*)d_in[7];   const float* b_rel = (const float*)d_in[8];
    const float* W_prob= (const float*)d_in[9];   const float* b_prob= (const float*)d_in[10];
    const float* W_so  = (const float*)d_in[11];  const float* W_oo  = (const float*)d_in[12];
    const float* W_rs  = (const float*)d_in[13];  const float* W_ro  = (const float*)d_in[14];
    const float* Wz = (const float*)d_in[15]; const float* Uz = (const float*)d_in[16]; const float* bz = (const float*)d_in[17];
    const float* Wr = (const float*)d_in[18]; const float* Ur = (const float*)d_in[19]; const float* br = (const float*)d_in[20];
    const float* Wh = (const float*)d_in[21]; const float* Uh = (const float*)d_in[22]; const float* bh = (const float*)d_in[23];
    const float* W_out_rel = (const float*)d_in[24]; const float* b_out_rel = (const float*)d_in[25];
    const float* W_cls_rel = (const float*)d_in[26]; const float* b_cls_rel = (const float*)d_in[27];
    const float* W_out_obj = (const float*)d_in[28]; const float* b_out_obj = (const float*)d_in[29];
    const float* W_cls_obj = (const float*)d_in[30]; const float* b_cls_obj = (const float*)d_in[31];

    // Output layout: obj_ref[P*NO], obj_preds[P], rel_logits[Q*NR], scpred[Q*NR], scent[P*NO]
    float* out        = (float*)d_out;
    float* o_obj_ref  = out;
    float* o_preds    = o_obj_ref + (long)Pc * NOc;
    float* o_rel_log  = o_preds + Pc;
    float* o_scpred   = o_rel_log + (long)Qc * NRc;
    float* o_scent    = o_scpred + (long)Qc * NRc;

    // Workspace layout (floats)
    float* ws = (float*)d_ws;
    long off = 0;
    auto alloc = [&](long n) { float* p = ws + off; off += n; return p; };
    float* obj_probs = alloc((long)Pc * NOc);
    float* xo = alloc((long)Pc * Hc);
    float* ho = alloc((long)Pc * Hc);
    float* mo = alloc((long)Pc * Hc);
    float* gzo = alloc((long)Pc * Hc);
    float* gro = alloc((long)Pc * Hc);
    float* gho = alloc((long)Pc * Hc);
    float* Ts  = alloc((long)Pc * Hc);   // doubles as segment-sum Ss
    float* To  = alloc((long)Pc * Hc);   // doubles as segment-sum So
    float* xr = alloc((long)Qc * Hc);
    float* hr = alloc((long)Qc * Hc);
    float* mr = alloc((long)Qc * Hc);
    float* gzr = alloc((long)Qc * Hc);
    float* grr = alloc((long)Qc * Hc);
    float* ghr = alloc((long)Qc * Hc);

    const long nO = (long)Pc * Hc;       // object-side elementwise size
    const long nR = (long)Qc * Hc;       // rel-side elementwise size
    const int EB = 256;
    auto eg = [&](long n) { return (unsigned)((n + EB - 1) / EB); };

    // ---- Input projections ----
    softmax_rows_kernel<<<Pc, 32, 0, stream>>>(obj_logits, obj_probs, Pc, NOc);
    launch_gemm(obj_fmaps, Dc, W_obj, Hc, b_obj, xo, Hc, Pc, Hc, Dc, 0, stream);
    launch_gemm(obj_probs, NOc, W_prob, Hc, b_prob, xo, Hc, Pc, Hc, NOc, 1, stream);
    launch_gemm(vr, Dc, W_rel, Hc, b_rel, xr, Hc, Qc, Hc, Dc, 0, stream);

    hipMemcpyAsync(ho, xo, nO * sizeof(float), hipMemcpyDeviceToDevice, stream);
    hipMemcpyAsync(hr, xr, nR * sizeof(float), hipMemcpyDeviceToDevice, stream);

    // ---- T GGNN timesteps ----
    for (int t = 0; t < Tc; ++t) {
        // mr = (ho@W_so)[s] + (ho@W_oo)[o]   (matmul-then-gather reordering)
        launch_gemm(ho, Hc, W_so, Hc, nullptr, Ts, Hc, Pc, Hc, Hc, 0, stream);
        launch_gemm(ho, Hc, W_oo, Hc, nullptr, To, Hc, Pc, Hc, Hc, 0, stream);
        gather_kernel<<<eg(nR), EB, 0, stream>>>(rel_inds, Ts, To, mr, Qc, Hc);

        // mo = segsum(hr,s)@W_rs + segsum(hr,o)@W_ro  (scatter-then-matmul reordering)
        zero_kernel<<<eg(nO), EB, 0, stream>>>(Ts, nO);
        zero_kernel<<<eg(nO), EB, 0, stream>>>(To, nO);
        scatter_kernel<<<eg(nR), EB, 0, stream>>>(rel_inds, hr, Ts, To, Qc, Hc);
        launch_gemm(Ts, Hc, W_rs, Hc, nullptr, mo, Hc, Pc, Hc, Hc, 0, stream);
        launch_gemm(To, Hc, W_ro, Hc, nullptr, mo, Hc, Pc, Hc, Hc, 1, stream);

        // Object GRU: ho = GRU(ho, mo)
        launch_gemm(mo, Hc, Wz, Hc, bz, gzo, Hc, Pc, Hc, Hc, 0, stream);
        launch_gemm(ho, Hc, Uz, Hc, nullptr, gzo, Hc, Pc, Hc, Hc, 1, stream);
        launch_gemm(mo, Hc, Wr, Hc, br, gro, Hc, Pc, Hc, Hc, 0, stream);
        launch_gemm(ho, Hc, Ur, Hc, nullptr, gro, Hc, Pc, Hc, Hc, 1, stream);
        gru_phase1_kernel<<<eg(nO), EB, 0, stream>>>(gzo, gro, ho, nO);
        launch_gemm(mo, Hc, Wh, Hc, bh, gho, Hc, Pc, Hc, Hc, 0, stream);
        launch_gemm(gro, Hc, Uh, Hc, nullptr, gho, Hc, Pc, Hc, Hc, 1, stream);
        gru_phase2_kernel<<<eg(nO), EB, 0, stream>>>(ho, gzo, gho, nO);

        // Relation GRU: hr = GRU(hr, mr)
        launch_gemm(mr, Hc, Wz, Hc, bz, gzr, Hc, Qc, Hc, Hc, 0, stream);
        launch_gemm(hr, Hc, Uz, Hc, nullptr, gzr, Hc, Qc, Hc, Hc, 1, stream);
        launch_gemm(mr, Hc, Wr, Hc, br, grr, Hc, Qc, Hc, Hc, 0, stream);
        launch_gemm(hr, Hc, Ur, Hc, nullptr, grr, Hc, Qc, Hc, Hc, 1, stream);
        gru_phase1_kernel<<<eg(nR), EB, 0, stream>>>(gzr, grr, hr, nR);
        launch_gemm(mr, Hc, Wh, Hc, bh, ghr, Hc, Qc, Hc, Hc, 0, stream);
        launch_gemm(grr, Hc, Uh, Hc, nullptr, ghr, Hc, Qc, Hc, Hc, 1, stream);
        gru_phase2_kernel<<<eg(nR), EB, 0, stream>>>(hr, gzr, ghr, nR);
    }

    // ---- Output heads (concat realized as two accumulated GEMMs) ----
    // rel: tanh(hr@Wout[:H] + xr@Wout[H:] + b) @ W_cls_rel + b_cls_rel
    launch_gemm(hr, Hc, W_out_rel, Hc, b_out_rel, mr, Hc, Qc, Hc, Hc, 0, stream);
    launch_gemm(xr, Hc, W_out_rel + (long)Hc * Hc, Hc, nullptr, mr, Hc, Qc, Hc, Hc, 1, stream);
    tanh_inplace_kernel<<<eg(nR), EB, 0, stream>>>(mr, nR);
    launch_gemm(mr, Hc, W_cls_rel, NRc, b_cls_rel, o_rel_log, NRc, Qc, NRc, Hc, 0, stream);

    // obj: tanh(ho@Wout[:H] + xo@Wout[H:] + b) @ W_cls_obj + b_cls_obj
    launch_gemm(ho, Hc, W_out_obj, Hc, b_out_obj, mo, Hc, Pc, Hc, Hc, 0, stream);
    launch_gemm(xo, Hc, W_out_obj + (long)Hc * Hc, Hc, nullptr, mo, Hc, Pc, Hc, Hc, 1, stream);
    tanh_inplace_kernel<<<eg(nO), EB, 0, stream>>>(mo, nO);
    launch_gemm(mo, Hc, W_cls_obj, NOc, b_cls_obj, o_obj_ref, NOc, Pc, NOc, Hc, 0, stream);

    // ---- Softmaxes + argmax preds ----
    softmax_rows_kernel<<<Qc, 32, 0, stream>>>(o_rel_log, o_scpred, Qc, NRc);
    softmax_rows_kernel<<<Pc, 32, 0, stream>>>(o_obj_ref, o_scent, Pc, NOc);
    argmax_kernel<<<Pc, 32, 0, stream>>>(o_obj_ref, o_preds, Pc, NOc);
}
